// MultiheadAttention_53498112639235
// MI455X (gfx1250) — matrix operand order repaired and checked by reading, not compiled, not run on hardware
//
#include <hip/hip_runtime.h>
#include <hip/hip_bf16.h>

#define DEVINL __device__ __forceinline__

typedef __attribute__((ext_vector_type(16))) __bf16 v16bf;
typedef __attribute__((ext_vector_type(8)))  float  v8f;
typedef __attribute__((ext_vector_type(4)))  unsigned int v4u;
typedef __attribute__((ext_vector_type(8)))  int v8i;
typedef __attribute__((ext_vector_type(4)))  int v4i;

constexpr int   T_SEQ   = 2048;
constexpr int   BATCH   = 2;
constexpr int   DMODEL  = 1024;
constexpr int   N_HEAD  = 16;
constexpr int   D_INNER = 64;
constexpr int   ROWS    = T_SEQ * BATCH;          // 4096
constexpr int   QKV_N   = 3 * N_HEAD * D_INNER;   // 3072
constexpr float SCALE_Q = 0.125f;                 // 1/sqrt(64)
constexpr float LN_EPS  = 1e-3f;

DEVINL unsigned short f32_to_bf16(float f) {
  union { float f; unsigned int u; } c; c.f = f;
  unsigned int u = c.u;
  u += 0x7FFFu + ((u >> 16) & 1u);   // round-to-nearest-even
  return (unsigned short)(u >> 16);
}

// ---------------- WMMA fragment loaders (bf16, wave32) -----------------
// A fragment: 16x32 (MxK), source row-major with leading dim ld.
// ISA 7.12.2: lane L -> m = L&15; elems 0..7 -> k = kb..kb+7,
// elems 8..15 -> k = 16+kb..16+kb+7 with kb = (L>=16)*8.
DEVINL v16bf load_a_frag(const unsigned short* base, int ld) {
  const int lane = threadIdx.x & 31;
  const int r    = lane & 15;
  const int kb   = (lane >> 4) * 8;
  union { uint4 u; __bf16 h[8]; } c0, c1;
  c0.u = *reinterpret_cast<const uint4*>(base + (size_t)r * ld + kb);
  c1.u = *reinterpret_cast<const uint4*>(base + (size_t)r * ld + 16 + kb);
  v16bf a;
#pragma unroll
  for (int i = 0; i < 8; ++i) { a[i] = c0.h[i]; a[8 + i] = c1.h[i]; }
  return a;
}

// B fragment: 32x16 (KxN), read from an N-major (N x K row-major) source.
// ISA 7.12.4/5: lane L -> n = L&15; elems 0..7 -> k = kb..kb+7,
// elems 8..15 -> k = 16+kb..16+kb+7 with kb = (L>=16)*8.
DEVINL v16bf load_b_frag(const unsigned short* base, int ld) {
  const int lane = threadIdx.x & 31;
  const int r    = lane & 15;          // n
  const int kb   = (lane >> 4) * 8;    // k base for this half-group
  union { uint4 u; __bf16 h[8]; } c0, c1;
  c0.u = *reinterpret_cast<const uint4*>(base + (size_t)r * ld + kb);
  c1.u = *reinterpret_cast<const uint4*>(base + (size_t)r * ld + 16 + kb);
  v16bf b;
#pragma unroll
  for (int i = 0; i < 8; ++i) { b[i] = c0.h[i]; b[8 + i] = c1.h[i]; }
  return b;
}

DEVINL v8f wmma_bf16(v16bf a, v16bf b, v8f c) {
  return __builtin_amdgcn_wmma_f32_16x16x32_bf16(
      /*neg_a=*/false, a, /*neg_b=*/false, b,
      /*c_mod=*/(short)0, c, /*reuse_a=*/false, /*reuse_b=*/false);
}

// ---------------- Tensor Data Mover: 2D tile global -> LDS --------------
// D# per CDNA5 ISA. data_size = 2 bytes (bf16).
// tile_w elements contiguous per line, tile_h lines, line stride
// stride_elems, tensor bounds tensor_w x tensor_h for OOB handling.
DEVINL void tdm_load_2d_bf16(unsigned lds_off, const void* gptr,
                             unsigned tile_w, unsigned tile_h,
                             unsigned tensor_w, unsigned tensor_h,
                             unsigned stride_elems) {
  unsigned long long ga = (unsigned long long)(uintptr_t)gptr;
  v4u g0;
  g0[0] = 1u;                                         // count=1 (valid), user mode
  g0[1] = lds_off;                                    // lds_addr (bytes)
  g0[2] = (unsigned)(ga & 0xFFFFFFFFu);               // global_addr[31:0]
  g0[3] = (unsigned)((ga >> 32) & 0x01FFFFFFu)        // global_addr[56:32]
          | (2u << 30);                               // type = 2 ("image")
  v8i g1;
  g1[0] = (int)(1u << 16);                            // wg_mask=0, data_size=1 (2B)
  g1[1] = (int)((tensor_w & 0xFFFFu) << 16);          // tensor_dim0[15:0]
  g1[2] = (int)(((tensor_w >> 16) & 0xFFFFu)          // tensor_dim0[31:16]
          | ((tensor_h & 0xFFFFu) << 16));            // tensor_dim1[15:0]
  g1[3] = (int)(((tensor_h >> 16) & 0xFFFFu)          // tensor_dim1[31:16]
          | ((tile_w & 0xFFFFu) << 16));              // tile_dim0
  g1[4] = (int)(tile_h & 0xFFFFu);                    // tile_dim1 (tile_dim2=0)
  g1[5] = (int)stride_elems;                          // tensor_dim0_stride[31:0]
  g1[6] = 0;                                          // stride hi / dim1_stride lo
  g1[7] = 0;
  v4i gz = {0, 0, 0, 0};
#if __clang_major__ >= 23
  v8i gz8 = {0, 0, 0, 0, 0, 0, 0, 0};
  __builtin_amdgcn_tensor_load_to_lds(g0, g1, gz, gz, gz8, 0);
#else
  __builtin_amdgcn_tensor_load_to_lds(g0, g1, gz, gz, 0);
#endif
}

// ---------------- stage 0: precision conversion / weight transpose ------
__global__ void cvt_f32_bf16(const float* __restrict__ x,
                             unsigned short* __restrict__ y, int n) {
  int i = blockIdx.x * blockDim.x + threadIdx.x;
  if (i < n) y[i] = f32_to_bf16(x[i]);
}

// w: (K x N) row-major  ->  wt: (N x K) row-major, bf16
__global__ void cvt_transpose(const float* __restrict__ w,
                              unsigned short* __restrict__ wt,
                              int Ncols, int Krows) {
  int i = blockIdx.x * blockDim.x + threadIdx.x;
  if (i < Ncols * Krows) {
    int n = i / Krows, k = i - n * Krows;
    wt[i] = f32_to_bf16(w[(size_t)k * Ncols + n]);
  }
}

// ---------------- stage 1: QKV projection GEMM --------------------------
__global__ void __launch_bounds__(128)
gemm_qkv(const unsigned short* __restrict__ Xbf,
         const unsigned short* __restrict__ Wt,
         unsigned short* __restrict__ Qbf,
         unsigned short* __restrict__ Kbf,
         unsigned short* __restrict__ Vtbf) {
  const int lane = threadIdx.x & 31;
  const int wave = threadIdx.x >> 5;
  const int m0 = blockIdx.x * 64 + wave * 16;
  const int n0 = blockIdx.y * 64;
  const int hi = lane >> 4;
  const int nn = lane & 15;

  v8f acc[4] = {};
  const unsigned short* Xrow = Xbf + (size_t)m0 * DMODEL;

  for (int kk = 0; kk < DMODEL; kk += 32) {
    v16bf a = load_a_frag(Xrow + kk, DMODEL);
#pragma unroll
    for (int nt = 0; nt < 4; ++nt) {
      v16bf b = load_b_frag(Wt + (size_t)(n0 + nt * 16) * DMODEL + kk, DMODEL);
      acc[nt] = wmma_bf16(a, b, acc[nt]);
    }
  }

#pragma unroll
  for (int nt = 0; nt < 4; ++nt) {
#pragma unroll
    for (int v = 0; v < 8; ++v) {
      int r = m0 + v + 8 * hi;       // X row = t*BATCH + b
      int t = r >> 1, b = r & 1;
      int c = n0 + nt * 16 + nn;     // col in [0,3072): s*1024 + h*64 + d
      int s = c >> 10;
      int rem = c & 1023;
      int h = rem >> 6, d = rem & 63;
      float val = acc[nt][v];
      size_t bh = (size_t)(b * N_HEAD + h);
      if (s == 0)      Qbf[(bh * T_SEQ + t) * D_INNER + d] = f32_to_bf16(val * SCALE_Q);
      else if (s == 1) Kbf[(bh * T_SEQ + t) * D_INNER + d] = f32_to_bf16(val);
      else             Vtbf[(bh * D_INNER + d) * T_SEQ + t] = f32_to_bf16(val);
    }
  }
}

// ---------------- stage 2: causal flash attention with TDM staging ------
// grid: (T/64, BATCH*N_HEAD), 128 threads (4 waves), wave owns 16 q-rows.
// K / V^T key blocks are staged into double-buffered LDS tiles by the
// Tensor Data Mover (one descriptor pair per block, issued by wave 0),
// overlapping the fetch of block n+1 with WMMA work on block n.
__global__ void __launch_bounds__(128)
attn_flash(const unsigned short* __restrict__ Qbf,
           const unsigned short* __restrict__ Kbf,
           const unsigned short* __restrict__ Vtbf,
           unsigned short* __restrict__ AVbf) {
  const int lane = threadIdx.x & 31;
  const int wave = threadIdx.x >> 5;
  const int bh   = blockIdx.y;             // b*N_HEAD + h
  const int b    = bh >> 4, h = bh & 15;
  const int i0   = blockIdx.x * 64 + wave * 16;
  const int hi   = lane >> 4;
  const int nn   = lane & 15;

  const unsigned short* Qb = Qbf + ((size_t)bh * T_SEQ + i0) * D_INNER;
  const unsigned short* Kb = Kbf + (size_t)bh * T_SEQ * D_INNER;
  const unsigned short* Vb = Vtbf + (size_t)bh * D_INNER * T_SEQ;

  const v16bf qa0 = load_a_frag(Qb + 0, D_INNER);
  const v16bf qa1 = load_a_frag(Qb + 32, D_INNER);

  v8f o[4] = {};
  float mrow[8], lrow[8];
#pragma unroll
  for (int v = 0; v < 8; ++v) { mrow[v] = -1e30f; lrow[v] = 0.0f; }

  __shared__ unsigned short kbuf[2][32][64];    // 8 KB: 32 keys x 64 d
  __shared__ unsigned short vbuf[2][64][32];    // 8 KB: 64 d x 32 keys
  __shared__ unsigned short ptile[4][16][32];   // 4 KB: wave-private P tiles

  const int nblk = blockIdx.x * 2 + 2;          // key blocks of 32 (uniform in WG)
  int cur = 0;

  if (wave == 0) {
    tdm_load_2d_bf16((unsigned)(size_t)&kbuf[0][0][0], Kb,
                     64, 32, D_INNER, T_SEQ, D_INNER);
    tdm_load_2d_bf16((unsigned)(size_t)&vbuf[0][0][0], Vb,
                     32, 64, T_SEQ, D_INNER, T_SEQ);
    __builtin_amdgcn_s_wait_tensorcnt(0);
  }
  __syncthreads();

  for (int blk = 0; blk < nblk; ++blk) {
    const int j0 = blk * 32;
    if (wave == 0 && blk + 1 < nblk) {          // prefetch next block via TDM
      const int jn = j0 + 32;
      tdm_load_2d_bf16((unsigned)(size_t)&kbuf[cur ^ 1][0][0],
                       Kb + (size_t)jn * D_INNER, 64, 32, D_INNER, T_SEQ, D_INNER);
      tdm_load_2d_bf16((unsigned)(size_t)&vbuf[cur ^ 1][0][0],
                       Vb + (size_t)jn, 32, 64, T_SEQ, D_INNER, T_SEQ);
    }

    // S = Q . K^T  (two 16x16 tiles over 32 keys), K read from LDS
    v16bf kb00 = load_b_frag(&kbuf[cur][0][0],  64);
    v16bf kb01 = load_b_frag(&kbuf[cur][0][32], 64);
    v16bf kb10 = load_b_frag(&kbuf[cur][16][0], 64);
    v16bf kb11 = load_b_frag(&kbuf[cur][16][32], 64);

    v8f s0 = {}, s1 = {};
    s0 = wmma_bf16(qa0, kb00, s0);
    s0 = wmma_bf16(qa1, kb01, s0);
    s1 = wmma_bf16(qa0, kb10, s1);
    s1 = wmma_bf16(qa1, kb11, s1);

    // online softmax update (row = v + 8*hi, col = nn within 16-lane group)
    float p0[8], p1[8];
#pragma unroll
    for (int v = 0; v < 8; ++v) {
      const int irow = i0 + v + 8 * hi;
      float a = s0[v]; if (j0 + nn > irow)      a = -1e30f;
      float c = s1[v]; if (j0 + 16 + nn > irow) c = -1e30f;
      float rm = fmaxf(a, c);
#pragma unroll
      for (int off = 1; off < 16; off <<= 1) rm = fmaxf(rm, __shfl_xor(rm, off, 16));
      const float mnew = fmaxf(mrow[v], rm);
      const float corr = __expf(mrow[v] - mnew);
      mrow[v] = mnew;
      const float e0 = __expf(a - mnew);
      const float e1 = __expf(c - mnew);
      float rs = e0 + e1;
#pragma unroll
      for (int off = 1; off < 16; off <<= 1) rs += __shfl_xor(rs, off, 16);
      lrow[v] = lrow[v] * corr + rs;
#pragma unroll
      for (int dt = 0; dt < 4; ++dt) o[dt][v] *= corr;
      p0[v] = e0; p1[v] = e1;
    }

    // restage P (C-frag layout) -> LDS -> A-frag layout (wave-private)
#pragma unroll
    for (int v = 0; v < 8; ++v) {
      const int m = v + 8 * hi;
      ptile[wave][m][nn]      = f32_to_bf16(p0[v]);
      ptile[wave][m][16 + nn] = f32_to_bf16(p1[v]);
    }
    const v16bf pa = load_a_frag(&ptile[wave][0][0], 32);

    // O += P . V   (V^T tile from LDS: rows d, contiguous keys)
#pragma unroll
    for (int dt = 0; dt < 4; ++dt) {
      v16bf vbf = load_b_frag(&vbuf[cur][dt * 16][0], 32);
      o[dt] = wmma_bf16(pa, vbf, o[dt]);
    }

    if (wave == 0) __builtin_amdgcn_s_wait_tensorcnt(0);
    __syncthreads();                            // TDM done + tile consumed
    cur ^= 1;
  }

  // normalize and emit attn_vec as bf16 rows (t*BATCH+b, h*64+d)
#pragma unroll
  for (int v = 0; v < 8; ++v) {
    const float inv = 1.0f / lrow[v];
    const int t = i0 + v + 8 * hi;
    const size_t rowbase = ((size_t)t * BATCH + b) * (N_HEAD * D_INNER) + h * D_INNER;
#pragma unroll
    for (int dt = 0; dt < 4; ++dt)
      AVbf[rowbase + dt * 16 + nn] = f32_to_bf16(o[dt][v] * inv);
  }
}

// ---------------- stage 3: output projection GEMM (fp32 out) ------------
__global__ void __launch_bounds__(128)
gemm_out(const unsigned short* __restrict__ AVbf,
         const unsigned short* __restrict__ Wt,
         float* __restrict__ AOf) {
  const int lane = threadIdx.x & 31;
  const int wave = threadIdx.x >> 5;
  const int m0 = blockIdx.x * 64 + wave * 16;
  const int n0 = blockIdx.y * 64;
  const int hi = lane >> 4;
  const int nn = lane & 15;

  v8f acc[4] = {};
  const unsigned short* Arow = AVbf + (size_t)m0 * DMODEL;

  for (int kk = 0; kk < DMODEL; kk += 32) {
    v16bf a = load_a_frag(Arow + kk, DMODEL);
#pragma unroll
    for (int nt = 0; nt < 4; ++nt) {
      v16bf b = load_b_frag(Wt + (size_t)(n0 + nt * 16) * DMODEL + kk, DMODEL);
      acc[nt] = wmma_bf16(a, b, acc[nt]);
    }
  }

#pragma unroll
  for (int nt = 0; nt < 4; ++nt) {
#pragma unroll
    for (int v = 0; v < 8; ++v) {
      int r = m0 + v + 8 * hi;
      int c = n0 + nt * 16 + nn;
      AOf[(size_t)r * DMODEL + c] = acc[nt][v];
    }
  }
}

// ---------------- stage 4: layer norm -----------------------------------
__global__ void __launch_bounds__(256)
layernorm_rows(const float* __restrict__ x,
               const float* __restrict__ gamma,
               const float* __restrict__ beta,
               float* __restrict__ out) {
  const int r = blockIdx.x;
  const float* row = x + (size_t)r * DMODEL;
  __shared__ float ssum[256], ssq[256];
  float s = 0.f, sq = 0.f;
  for (int c = threadIdx.x; c < DMODEL; c += 256) {
    float v = row[c]; s += v; sq += v * v;
  }
  ssum[threadIdx.x] = s; ssq[threadIdx.x] = sq;
  __syncthreads();
  for (int off = 128; off > 0; off >>= 1) {
    if ((int)threadIdx.x < off) {
      ssum[threadIdx.x] += ssum[threadIdx.x + off];
      ssq[threadIdx.x]  += ssq[threadIdx.x + off];
    }
    __syncthreads();
  }
  const float mean = ssum[0] * (1.0f / DMODEL);
  const float var  = ssq[0] * (1.0f / DMODEL) - mean * mean;
  const float rstd = rsqrtf(var + LN_EPS);
  for (int c = threadIdx.x; c < DMODEL; c += 256)
    out[(size_t)r * DMODEL + c] = (row[c] - mean) * rstd * gamma[c] + beta[c];
}

// ---------------- host launcher -----------------------------------------
extern "C" void kernel_launch(void* const* d_in, const int* in_sizes, int n_in,
                              void* d_out, int out_size, void* d_ws, size_t ws_size,
                              hipStream_t stream) {
  const float* inps  = (const float*)d_in[0];   // (T,B,1024)
  const float* Wqkv  = (const float*)d_in[1];   // (1024,3072)
  const float* Wout  = (const float*)d_in[2];   // (1024,1024)
  const float* gamma = (const float*)d_in[3];
  const float* beta  = (const float*)d_in[4];
  // d_in[5] attn_mask: causal triu(k=1), computed analytically in-kernel.
  float* out = (float*)d_out;

  // workspace layout (64 MB total)
  unsigned short* Xbf   = (unsigned short*)d_ws;                 // 4096x1024
  unsigned short* WqkvT = Xbf   + (size_t)ROWS * DMODEL;         // 3072x1024 (N-major)
  unsigned short* WoutT = WqkvT + (size_t)QKV_N * DMODEL;        // 1024x1024 (N-major)
  unsigned short* Qbf   = WoutT + (size_t)DMODEL * DMODEL;       // (B*H,T,64), pre-scaled
  unsigned short* Kbf   = Qbf   + (size_t)BATCH * N_HEAD * T_SEQ * D_INNER;
  unsigned short* Vtbf  = Kbf   + (size_t)BATCH * N_HEAD * T_SEQ * D_INNER; // (B*H,64,T)
  unsigned short* AVbf  = Vtbf  + (size_t)BATCH * N_HEAD * T_SEQ * D_INNER; // 4096x1024
  float*          AOf   = (float*)(AVbf + (size_t)ROWS * DMODEL);           // 4096x1024 f32

  const int nX = ROWS * DMODEL;
  cvt_f32_bf16<<<(nX + 255) / 256, 256, 0, stream>>>(inps, Xbf, nX);
  cvt_transpose<<<(QKV_N * DMODEL + 255) / 256, 256, 0, stream>>>(Wqkv, WqkvT, QKV_N, DMODEL);
  cvt_transpose<<<(DMODEL * DMODEL + 255) / 256, 256, 0, stream>>>(Wout, WoutT, DMODEL, DMODEL);

  gemm_qkv<<<dim3(ROWS / 64, QKV_N / 64), 128, 0, stream>>>(Xbf, WqkvT, Qbf, Kbf, Vtbf);
  attn_flash<<<dim3(T_SEQ / 64, BATCH * N_HEAD), 128, 0, stream>>>(Qbf, Kbf, Vtbf, AVbf);
  gemm_out<<<dim3(ROWS / 64, DMODEL / 64), 128, 0, stream>>>(AVbf, WoutT, AOf);
  layernorm_rows<<<ROWS, 256, 0, stream>>>(AOf, gamma, beta, out);

  (void)in_sizes; (void)n_in; (void)out_size; (void)ws_size;
}
